// GatedGraphX_30537217475304
// MI455X (gfx1250) — compile-verified, matrix operand order checked by reading, not run on hardware
//
#include <hip/hip_runtime.h>
#include <hip/hip_bf16.h>

typedef __bf16 bf16;
typedef __attribute__((ext_vector_type(16))) __bf16 bf16x16;
typedef __attribute__((ext_vector_type(4)))  __bf16 bf16x4;
typedef __attribute__((ext_vector_type(8)))  float  f32x8;

constexpr int NN   = 4096;   // nodes
constexpr int HH   = 64;     // hidden
constexpr int EE   = 10;     // edge types
constexpr int C3   = 192;    // 3*H
constexpr int ITER = 10;

// WMMA bf16 16x16x32 fragment K-mapping (A: row=lane&15; B: col=lane&15):
// lane<16  -> K in {0..7} then {16..23}
// lane>=16 -> K in {8..15} then {24..31}
__device__ __host__ inline int kmap(int lane, int idx) {
    int kh = (lane >> 4) * 8;
    return (idx < 8) ? (kh + idx) : (16 + kh + (idx - 8));
}

__device__ inline f32x8 zero8() {
    f32x8 z = {0.f, 0.f, 0.f, 0.f, 0.f, 0.f, 0.f, 0.f};
    return z;
}

// A fragment from a row-major matrix (used by kP on act, ld = 64):
// both 8-element chunks are contiguous -> 2x b128 loads.
__device__ inline bf16x16 load_frag_a(const bf16* base, int ld, int lane) {
    int row = lane & 15;
    int kh  = (lane >> 4) * 8;
    const bf16* p = base + row * ld + kh;
    bf16x16 f;
#pragma unroll
    for (int i = 0; i < 8; ++i) f[i] = p[i];
#pragma unroll
    for (int i = 0; i < 8; ++i) f[8 + i] = p[16 + i];
    return f;
}

__device__ inline f32x8 wmma_bf16(bf16x16 a, bf16x16 b, f32x8 c) {
    return __builtin_amdgcn_wmma_f32_16x16x32_bf16(false, a, false, b,
                                                   (short)0, c, false, false);
}

// ---- prep: transpose+convert edge[e][j][i] f32 into A-fragment-packed bf16:
// edgeP[e][mtile(256)][kblock(128)][lane(32)][16]  (i = M dim, j = K dim)
__global__ __launch_bounds__(256) void kTrans(const float* __restrict__ edge,
                                              bf16* __restrict__ edgeP) {
    __shared__ float tile[32][33];
    int e  = blockIdx.z;
    int bi = blockIdx.x * 32;          // i tile (M rows)
    int bj = blockIdx.y * 32;          // j tile (K cols)
    int tid = threadIdx.x;
    int tx = tid & 31, ty = tid >> 5;  // ty in 0..7
    const float* src = edge + (size_t)e * NN * NN;
#pragma unroll
    for (int r = 0; r < 4; ++r) {
        int j = bj + ty + r * 8;
        int i = bi + tx;
        tile[ty + r * 8][tx] = src[(size_t)j * NN + i];  // tile[K_local][M_local]
    }
    __syncthreads();
    // repack: 2 sub-mtiles x 32 lanes x 4 quads of 4 bf16
    int c = tid >> 7;          // sub-mtile 0..1
    int l = (tid >> 2) & 31;   // lane
    int q = tid & 3;           // quad
    int r = l & 15;            // fragment row within mtile
    bf16x4 v;
#pragma unroll
    for (int u = 0; u < 4; ++u) {
        int k = kmap(l, q * 4 + u);
        v[u] = (bf16)tile[k][c * 16 + r];
    }
    size_t mt = (size_t)(bi >> 4) + c;
    bf16* dst = edgeP + (((size_t)e * 256 + mt) * 128 + (bj >> 5)) * 512 +
                l * 16 + q * 4;
    *(bf16x4*)dst = v;
}

// ---- per-iter: pack h (f32) into B-fragment bf16: hP[kblock(128)][nt(4)][lane(32)][16]
__global__ void kPackH(const float* __restrict__ h, bf16* __restrict__ hP) {
    int id = blockIdx.x * 256 + threadIdx.x;  // 128*4*32 = 16384
    int l  = id & 31;
    int nt = (id >> 5) & 3;
    int kb = id >> 7;
    int col = nt * 16 + (l & 15);
    bf16x16 v;
#pragma unroll
    for (int idx = 0; idx < 16; ++idx) {
        int k = kmap(l, idx);
        v[idx] = (bf16)h[(size_t)(kb * 32 + k) * HH + col];
    }
    *(bf16x16*)(hP + (size_t)id * 16) = v;
}

// ---- prep: pack wz_wr_wh (f32) into B-fragment bf16:
// WP[e][kkb(2)][nt(12)][lane(32)][16]
__global__ void kWpack(const float* __restrict__ W, bf16* __restrict__ WP) {
    int id = blockIdx.x * 256 + threadIdx.x;  // 10*2*12*32 = 7680
    int l   = id & 31;
    int nt  = (id >> 5) % 12;
    int kkb = (id / (12 * 32)) & 1;
    int e   = id / (2 * 12 * 32);
    int col = nt * 16 + (l & 15);
    bf16x16 v;
#pragma unroll
    for (int idx = 0; idx < 16; ++idx) {
        int k = kmap(l, idx);
        v[idx] = (bf16)W[((size_t)e * HH + kkb * 32 + k) * C3 + col];
    }
    *(bf16x16*)(WP + (size_t)id * 16) = v;
}

// ---- prep: baW[c] = sum_e sum_k ba[k] * W[e][k][c]
__global__ void kBaW(const float* __restrict__ ba, const float* __restrict__ W,
                     float* __restrict__ baW) {
    int c = threadIdx.x;
    float s = 0.f;
    for (int e = 0; e < EE; ++e)
        for (int k = 0; k < HH; ++k)
            s += ba[k] * W[((size_t)e * HH + k) * C3 + c];
    baW[c] = s;
}

// ---- prep: inpW = input @ input_wzrh
__global__ void kInpW(const float* __restrict__ input, const float* __restrict__ inwz,
                      float* __restrict__ inpW) {
    int idx = blockIdx.x * 256 + threadIdx.x;
    int n = idx / C3, c = idx % C3;
    float s = 0.f;
    for (int k = 0; k < HH; ++k) s += input[n * HH + k] * inwz[k * C3 + c];
    inpW[idx] = s;
}

// ---- heavy GEMM: act[e] (4096x64) = edgeT[e] @ h.  A from packed edgeP (linear
// HBM stream, 2x b128/frag), B from packed hP (L2-resident, 2x b128/frag).
// 256 threads = 8 waves; wave computes 32(M) x 64(N); block covers 256 M rows.
__global__ __launch_bounds__(256) void kM(const bf16* __restrict__ edgeP,
                                          const bf16* __restrict__ hP,
                                          bf16* __restrict__ act) {
    int e = blockIdx.y;
    int tid = threadIdx.x, wave = tid >> 5, lane = tid & 31;
    int mt0 = blockIdx.x * 16 + wave * 2;  // 2 mtiles per wave

    const bf16* a0 = edgeP + (((size_t)e * 256 + mt0) * 128) * 512 + lane * 16;
    const bf16* a1 = a0 + (size_t)128 * 512;
    const bf16* bb = hP + (size_t)lane * 16;

    f32x8 acc[2][4];
#pragma unroll
    for (int m = 0; m < 2; ++m)
#pragma unroll
        for (int nt = 0; nt < 4; ++nt) acc[m][nt] = zero8();

    for (int kb = 0; kb < 128; ++kb) {
        bf16x16 af0 = *(const bf16x16*)(a0 + (size_t)kb * 512);
        bf16x16 af1 = *(const bf16x16*)(a1 + (size_t)kb * 512);
        __builtin_prefetch(a0 + (size_t)(kb + 16) * 512, 0, 1);
#pragma unroll
        for (int nt = 0; nt < 4; ++nt) {
            bf16x16 bfr = *(const bf16x16*)(bb + (size_t)(kb * 4 + nt) * 512);
            acc[0][nt] = wmma_bf16(af0, bfr, acc[0][nt]);
            acc[1][nt] = wmma_bf16(af1, bfr, acc[1][nt]);
        }
    }

    int colb = lane & 15, rowh = (lane >> 4) * 8;
#pragma unroll
    for (int m = 0; m < 2; ++m)
#pragma unroll
        for (int nt = 0; nt < 4; ++nt)
#pragma unroll
            for (int r = 0; r < 8; ++r) {
                int grow = (mt0 + m) * 16 + rowh + r;
                act[((size_t)e * NN + grow) * HH + nt * 16 + colb] =
                    (bf16)acc[m][nt][r];
            }
}

// ---- projection: S (4096x192) = sum_e act[e] @ W[e] + 10*inpW + baW
__global__ __launch_bounds__(256) void kP(const bf16* __restrict__ act,
                                          const bf16* __restrict__ WP,
                                          const float* __restrict__ inpW,
                                          const float* __restrict__ baW,
                                          float* __restrict__ S) {
    int wave = threadIdx.x >> 5, lane = threadIdx.x & 31;
    int mrow = blockIdx.x * 128 + wave * 16;

    f32x8 acc[12];
#pragma unroll
    for (int nt = 0; nt < 12; ++nt) acc[nt] = zero8();

    for (int e = 0; e < EE; ++e) {
        const bf16* Abase = act + ((size_t)e * NN + mrow) * HH;
#pragma unroll
        for (int kkb = 0; kkb < 2; ++kkb) {
            bf16x16 af = load_frag_a(Abase + kkb * 32, HH, lane);
            const bf16* bb = WP + ((size_t)(e * 2 + kkb) * 12) * 512 + lane * 16;
#pragma unroll
            for (int nt = 0; nt < 12; ++nt) {
                bf16x16 bfr = *(const bf16x16*)(bb + (size_t)nt * 512);
                acc[nt] = wmma_bf16(af, bfr, acc[nt]);
            }
        }
    }

    int colb = lane & 15, rowh = (lane >> 4) * 8;
#pragma unroll
    for (int nt = 0; nt < 12; ++nt)
#pragma unroll
        for (int r = 0; r < 8; ++r) {
            int grow = mrow + rowh + r;
            int gcol = nt * 16 + colb;
            S[(size_t)grow * C3 + gcol] =
                acc[nt][r] + (float)EE * inpW[(size_t)grow * C3 + gcol] + baW[gcol];
        }
}

// ---- GRU-style gate update
__global__ __launch_bounds__(256) void kGate(const float* __restrict__ hin,
                                             const float* __restrict__ S,
                                             const float* __restrict__ uzur,
                                             const float* __restrict__ uhm,
                                             float* __restrict__ hout) {
    __shared__ float sh_h[4][64];
    __shared__ float sh_rh[4][64];
    int tid = threadIdx.x;
    int ln = tid >> 6;
    int c  = tid & 63;
    int n  = blockIdx.x * 4 + ln;

    float hv = hin[(size_t)n * HH + c];
    sh_h[ln][c] = hv;
    __syncthreads();

    float uz = 0.f, ur = 0.f;
#pragma unroll 8
    for (int k = 0; k < HH; ++k) {
        float hk = sh_h[ln][k];
        uz += hk * uzur[k * 128 + c];
        ur += hk * uzur[k * 128 + 64 + c];
    }
    const float* Sn = S + (size_t)n * C3;
    float z = 1.f / (1.f + __expf(-(Sn[c] + uz)));
    float r = 1.f / (1.f + __expf(-(Sn[64 + c] + ur)));

    sh_rh[ln][c] = r * hv;
    __syncthreads();

    float uhv = 0.f;
#pragma unroll 8
    for (int k = 0; k < HH; ++k) uhv += sh_rh[ln][k] * uhm[k * 64 + c];

    float ht = tanhf(Sn[128 + c] + uhv);
    hout[(size_t)n * HH + c] = (1.f - z) * hv + z * ht;
}

extern "C" void kernel_launch(void* const* d_in, const int* in_sizes, int n_in,
                              void* d_out, int out_size, void* d_ws, size_t ws_size,
                              hipStream_t stream) {
    (void)in_sizes; (void)n_in; (void)out_size; (void)ws_size;
    const float* input  = (const float*)d_in[0];
    const float* hidden = (const float*)d_in[1];
    const float* edge   = (const float*)d_in[2];
    const float* ba     = (const float*)d_in[3];
    const float* W      = (const float*)d_in[4];
    const float* uzur   = (const float*)d_in[5];
    const float* uhm    = (const float*)d_in[6];
    const float* inwz   = (const float*)d_in[7];
    // d_in[8] = iteration (device scalar); fixed at 10 (graph-capture safe).
    float* out = (float*)d_out;

    char* ws = (char*)d_ws;
    size_t off = 0;
    auto carve = [&](size_t bytes) -> char* {
        char* p = ws + off;
        off += (bytes + 255) & ~(size_t)255;
        return p;
    };
    bf16*  edgeP = (bf16*)carve((size_t)EE * NN * NN * sizeof(bf16));  // 335.5 MB
    bf16*  actb  = (bf16*)carve((size_t)EE * NN * HH * sizeof(bf16));  // 5.2 MB
    bf16*  WP    = (bf16*)carve((size_t)EE * HH * C3 * sizeof(bf16));
    bf16*  hP    = (bf16*)carve((size_t)NN * HH * sizeof(bf16));       // 512 KB
    float* inpW  = (float*)carve((size_t)NN * C3 * sizeof(float));
    float* baW   = (float*)carve(1024);
    float* S     = (float*)carve((size_t)NN * C3 * sizeof(float));
    float* hA    = (float*)carve((size_t)NN * HH * sizeof(float));
    float* hB    = (float*)carve((size_t)NN * HH * sizeof(float));

    // one-time prep
    kTrans<<<dim3(NN / 32, NN / 32, EE), 256, 0, stream>>>(edge, edgeP);
    kWpack<<<(EE * 2 * 12 * 32) / 256, 256, 0, stream>>>(W, WP);
    kBaW<<<1, C3, 0, stream>>>(ba, W, baW);
    kInpW<<<(NN * C3) / 256, 256, 0, stream>>>(input, inwz, inpW);

    for (int it = 0; it < ITER; ++it) {
        const float* cur = (it == 0) ? hidden : ((it & 1) ? hA : hB);
        float* nxt = (it == ITER - 1) ? out : ((it & 1) ? hB : hA);
        kPackH<<<(128 * 4 * 32) / 256, 256, 0, stream>>>(cur, hP);
        kM<<<dim3(NN / 256, EE), 256, 0, stream>>>(edgeP, hP, actb);
        kP<<<NN / 128, 256, 0, stream>>>(actb, WP, inpW, baW, S);
        kGate<<<NN / 4, 256, 0, stream>>>(cur, S, uzur, uhm, nxt);
    }
}
